// SparseMoELayer_50294067036403
// MI455X (gfx1250) — matrix operand order repaired and checked by reading, not compiled, not run on hardware
//
#include <hip/hip_runtime.h>
#include <hip/hip_bf16.h>
#include <math.h>

#define D_IN   1024
#define D_OUT  1024
#define N_EXP  8

typedef __attribute__((ext_vector_type(16))) __bf16 v16bf;
typedef __attribute__((ext_vector_type(8)))  float  v8f;

// ---------------------------------------------------------------------------
// Kernel 1: gating. One wave32 per token. Computes dense per-token expert
// weights w[t][e] (softmax over top-2 logits, 0 elsewhere) into d_ws.
// ---------------------------------------------------------------------------
__global__ __launch_bounds__(256) void moe_gate_kernel(
    const float* __restrict__ x, const float* __restrict__ Wg,
    const float* __restrict__ bg, float* __restrict__ wdense)
{
  const int wave = threadIdx.x >> 5;
  const int lane = threadIdx.x & 31;
  const int t    = blockIdx.x * 8 + wave;
  const float* xt = x + (size_t)t * D_IN;

  float acc[N_EXP];
#pragma unroll
  for (int e = 0; e < N_EXP; ++e) acc[e] = 0.f;

#pragma unroll
  for (int it = 0; it < D_IN / 128; ++it) {
    const int k = it * 128 + lane * 4;
    const float4 xv = *(const float4*)(xt + k);
#pragma unroll
    for (int e = 0; e < N_EXP; ++e) {
      const float4 gv = *(const float4*)(Wg + e * D_IN + k);
      acc[e] += xv.x * gv.x + xv.y * gv.y + xv.z * gv.z + xv.w * gv.w;
    }
  }

#pragma unroll
  for (int e = 0; e < N_EXP; ++e) {
    float v = acc[e];
#pragma unroll
    for (int off = 16; off > 0; off >>= 1) v += __shfl_xor(v, off, 32);
    acc[e] = v + bg[e];
  }

  if (lane == 0) {
    int i1 = 0; float v1 = acc[0];
#pragma unroll
    for (int e = 1; e < N_EXP; ++e) if (acc[e] > v1) { v1 = acc[e]; i1 = e; }
    int i2 = -1; float v2 = -3.4e38f;
#pragma unroll
    for (int e = 0; e < N_EXP; ++e) if (e != i1 && acc[e] > v2) { v2 = acc[e]; i2 = e; }
    const float e2 = __expf(v2 - v1);          // softmax over {v1, v2}
    const float inv = 1.f / (1.f + e2);
    const float w1 = inv, w2 = e2 * inv;
#pragma unroll
    for (int e = 0; e < N_EXP; ++e) {
      float w = (e == i1) ? w1 : ((e == i2) ? w2 : 0.f);
      wdense[(size_t)t * N_EXP + e] = w;
    }
  }
}

// ---------------------------------------------------------------------------
// Kernel 2: dense-over-experts GEMM with gate weighting.
// Block: 256 threads (8 waves). Block tile 128(M) x 64(N).
// Each wave: 16(M) x 64(N) strip = 4x v_wmma_f32_16x16x32_bf16 accumulators.
// W tiles (64 cols x 32 K, f32) are copied global->LDS with
// GLOBAL_LOAD_ASYNC_TO_LDS_B128, double-buffered and pipelined across the
// whole flattened (expert x K-chunk) loop; ASYNCcnt gates consumption.
// The LDS operand of the async asm is derived from &ldsW so the shared
// array's address escapes into the asm (keeps GlobalsAA honest about the
// asm writing LDS) and its low 32 bits are the LDS byte offset.
// ---------------------------------------------------------------------------
#define WROWS    64
#define WSTRIDE  36                    // f32 per LDS row (144B, padded)
#define NCHUNKS  (D_IN / 32)           // 32 K-chunks per expert
#define TOTCH    (N_EXP * NCHUNKS)     // 256 chunks total

__global__ __launch_bounds__(256) void moe_expert_kernel(
    const float* __restrict__ x, const float* __restrict__ We,
    const float* __restrict__ be, const float* __restrict__ wdense,
    float* __restrict__ out)
{
  __shared__ float ldsW[2][WROWS * WSTRIDE];

  const int tid  = threadIdx.x;
  const int wave = tid >> 5;
  const int lane = tid & 31;
  const int hi   = lane >> 4;     // 0: lanes 0-15, 1: lanes 16-31
  const int l15  = lane & 15;

  const int n0 = blockIdx.x * 64;               // output-feature base of block
  const int m0 = blockIdx.y * 128 + wave * 16;  // token base of this wave

  const float* xrow = x + (size_t)(m0 + l15) * D_IN;  // A row for this lane
  const int koff = hi * 8;  // lanes>=16 hold K {8..15, 24..31}

  // staging coords: thread -> (column, K-quarter of a 32-wide chunk)
  const int sc = tid >> 2;          // 0..63 column within N tile
  const int sk = (tid & 3) * 8;     // 0,8,16,24

  // issue one chunk's async global->LDS copies (32B per thread, 2 x b128).
  // LDS offset operand comes from the real address of ldsW (ptrtoint escape).
  auto issue_chunk = [&](int c) {
    const int ce  = c >> 5;              // expert
    const int ck  = (c & 31) * 32;       // k0
    const float* wp = We + ((size_t)ce << 20) + (size_t)(n0 + sc) * D_IN + ck + sk;
    const uint32_t lo =
        (uint32_t)(uintptr_t)&ldsW[c & 1][sc * WSTRIDE + sk];
    asm volatile("global_load_async_to_lds_b128 %0, %1, off"
                 :: "v"(lo), "v"(wp) : "memory");
    asm volatile("global_load_async_to_lds_b128 %0, %1, off"
                 :: "v"(lo + 16u), "v"(wp + 4) : "memory");
  };

  v8f outacc[4];
#pragma unroll
  for (int j = 0; j < 4; ++j) outacc[j] = (v8f)0.f;

  // pipeline prologue: chunks 0 and 1 in flight
  issue_chunk(0);
  issue_chunk(1);

  int c = 0;
  for (int e = 0; e < N_EXP; ++e) {
    // per-lane gate weights for the 8 token rows this lane touches in C
    float wv[8];
#pragma unroll
    for (int r = 0; r < 8; ++r)
      wv[r] = wdense[(size_t)(m0 + r + hi * 8) * N_EXP + e];

    float bias[4];
#pragma unroll
    for (int j = 0; j < 4; ++j)
      bias[j] = be[e * D_OUT + n0 + j * 16 + l15];

    v8f acc[4];
#pragma unroll
    for (int j = 0; j < 4; ++j) acc[j] = (v8f)0.f;

    for (int kc = 0; kc < NCHUNKS; ++kc, ++c) {
      // retire chunk c (oldest 2 async ops); keep chunk c+1 in flight
      if (c + 1 < TOTCH) {
        asm volatile("s_wait_asynccnt 0x2" ::: "memory");
      } else {
        asm volatile("s_wait_asynccnt 0x0" ::: "memory");
      }
      __syncthreads();

      const int k0 = kc * 32;

      // ---- A fragment from global, f32 -> bf16 ----
      const float* ap = xrow + k0 + koff;
      const float4 a0 = *(const float4*)(ap);
      const float4 a1 = *(const float4*)(ap + 4);
      const float4 a2 = *(const float4*)(ap + 16);
      const float4 a3 = *(const float4*)(ap + 20);
      __builtin_prefetch((const void*)(ap + 32), 0, 1);
      v16bf af;
      af[0]  = (__bf16)a0.x; af[1]  = (__bf16)a0.y;
      af[2]  = (__bf16)a0.z; af[3]  = (__bf16)a0.w;
      af[4]  = (__bf16)a1.x; af[5]  = (__bf16)a1.y;
      af[6]  = (__bf16)a1.z; af[7]  = (__bf16)a1.w;
      af[8]  = (__bf16)a2.x; af[9]  = (__bf16)a2.y;
      af[10] = (__bf16)a2.z; af[11] = (__bf16)a2.w;
      af[12] = (__bf16)a3.x; af[13] = (__bf16)a3.y;
      af[14] = (__bf16)a3.z; af[15] = (__bf16)a3.w;

      // ---- B fragments from LDS (f32) -> bf16, then WMMA ----
      const float* bbase = &ldsW[c & 1][0];
#pragma unroll
      for (int j = 0; j < 4; ++j) {
        const float* bp = bbase + (j * 16 + l15) * WSTRIDE + hi * 8;
        const float4 b0 = *(const float4*)(bp);
        const float4 b1 = *(const float4*)(bp + 4);
        const float4 b2 = *(const float4*)(bp + 16);
        const float4 b3 = *(const float4*)(bp + 20);
        v16bf bfr;
        bfr[0]  = (__bf16)b0.x; bfr[1]  = (__bf16)b0.y;
        bfr[2]  = (__bf16)b0.z; bfr[3]  = (__bf16)b0.w;
        bfr[4]  = (__bf16)b1.x; bfr[5]  = (__bf16)b1.y;
        bfr[6]  = (__bf16)b1.z; bfr[7]  = (__bf16)b1.w;
        bfr[8]  = (__bf16)b2.x; bfr[9]  = (__bf16)b2.y;
        bfr[10] = (__bf16)b2.z; bfr[11] = (__bf16)b2.w;
        bfr[12] = (__bf16)b3.x; bfr[13] = (__bf16)b3.y;
        bfr[14] = (__bf16)b3.z; bfr[15] = (__bf16)b3.w;
        acc[j] = __builtin_amdgcn_wmma_f32_16x16x32_bf16(
            false, af, false, bfr, (short)0, acc[j], false, false);
      }
      __syncthreads();  // all waves done reading buf[c&1]

      if (c + 2 < TOTCH) issue_chunk(c + 2);  // refill freed buffer
    }

    // ---- apply bias + gate weight, accumulate into output regs ----
#pragma unroll
    for (int j = 0; j < 4; ++j)
#pragma unroll
      for (int r = 0; r < 8; ++r)
        outacc[j][r] += wv[r] * (acc[j][r] + bias[j]);
  }

  // ---- store: C layout VGPR r -> M = r + 8*hi, lane&15 -> N ----
#pragma unroll
  for (int j = 0; j < 4; ++j) {
    const int n = n0 + j * 16 + l15;
#pragma unroll
    for (int r = 0; r < 8; ++r) {
      const int m = m0 + r + hi * 8;
      out[(size_t)m * D_OUT + n] = outacc[j][r];
    }
  }
}

// ---------------------------------------------------------------------------
extern "C" void kernel_launch(void* const* d_in, const int* in_sizes, int n_in,
                              void* d_out, int out_size, void* d_ws, size_t ws_size,
                              hipStream_t stream) {
  const float* x   = (const float*)d_in[0];  // [4,2048,1024]
  const float* We  = (const float*)d_in[1];  // [8,1024,1024]
  const float* be  = (const float*)d_in[2];  // [8,1024]
  const float* Wg  = (const float*)d_in[3];  // [8,1024]
  const float* bg  = (const float*)d_in[4];  // [8]
  float* out = (float*)d_out;                // [T,1024]
  float* wdense = (float*)d_ws;              // [T,8] scratch

  const int T = in_sizes[0] / D_IN;          // 8192 tokens

  moe_gate_kernel<<<T / 8, 256, 0, stream>>>(x, Wg, bg, wdense);
  moe_expert_kernel<<<dim3(D_OUT / 64, T / 128), 256, 0, stream>>>(
      x, We, be, wdense, out);
}
